// MultiHeadAttention_79714593013958
// MI455X (gfx1250) — matrix operand order repaired and checked by reading, not compiled, not run on hardware
//
#include <hip/hip_runtime.h>

typedef __bf16 bf16;
typedef __attribute__((ext_vector_type(16))) __bf16 v16bf;
typedef __attribute__((ext_vector_type(8)))  __bf16 v8bf;
typedef __attribute__((ext_vector_type(8)))  float  v8f;

union ABFrag { v16bf v; v8bf h[2]; };

#define NEG_BIG (-1.0e30f)

__device__ __forceinline__ bf16 f2bf(float f) {
  unsigned u = __builtin_bit_cast(unsigned, f);
  unsigned r = (u + 0x7FFFu + ((u >> 16) & 1u)) >> 16;
  unsigned short s = (unsigned short)r;
  return __builtin_bit_cast(bf16, s);
}
__device__ __forceinline__ float bf2f(bf16 b) {
  unsigned short s = __builtin_bit_cast(unsigned short, b);
  unsigned u = ((unsigned)s) << 16;
  return __builtin_bit_cast(float, u);
}
__device__ __forceinline__ v8f wmma_bf16(v16bf a, v16bf b, v8f c) {
  // D = A(16x32 bf16) * B(32x16 bf16) + C(16x16 f32)
  return __builtin_amdgcn_wmma_f32_16x16x32_bf16(
      /*neg_a=*/false, a, /*neg_b=*/false, b,
      /*c_mod=*/(short)0, c, /*reuse_a=*/false, /*reuse_b=*/false);
}

// ---------------------------------------------------------------------------
// fp32 -> bf16 elementwise convert
// ---------------------------------------------------------------------------
__global__ void cvt_f32_to_bf16(const float* __restrict__ src,
                                bf16* __restrict__ dst, long n) {
  long i = (long)blockIdx.x * blockDim.x + threadIdx.x;
  long stride = (long)gridDim.x * blockDim.x;
  for (; i < n; i += stride) dst[i] = f2bf(src[i]);
}

// ---------------------------------------------------------------------------
// C = A(MxK) @ W^T(NxK) + bias, bf16 inputs, fp32 accumulate via WMMA.
// Block = 256 threads = 8 waves; block tile 128(M) x 256(N); wave tile 64x64.
// mode 0: out bf16, scatter [b,h,s,d]   (Q / K raw)
// mode 1: out bf16, scatter [b,h,d,s]   (V transposed)
// mode 2: out fp32, row-major [M,N]     (final projection)
// Hardcodes S=2048 rows/batch, DPH=128, H=16 (fixed problem sizes).
// ---------------------------------------------------------------------------
__global__ void __launch_bounds__(256)
gemm_wmma_bf16(const bf16* __restrict__ A, const bf16* __restrict__ W,
               const float* __restrict__ bias, void* __restrict__ outp,
               int M, int N, int K, int mode) {
  const int lane = threadIdx.x & 31;
  const int wid  = threadIdx.x >> 5;
  const int wm   = (wid >> 2) << 6;   // 0 / 64
  const int wn   = (wid & 3) << 6;    // 0,64,128,192
  const long tileM = (long)blockIdx.y * 128 + wm;
  const long tileN = (long)blockIdx.x * 256 + wn;
  const int half = lane >> 4;
  const int l16  = lane & 15;
  const int kbA  = half << 4;   // A-frag K base within 32
  const int kbB  = half << 4;   // B-frag K base within 32

  v8f acc[4][4];
  const v8f vz = {};
#pragma unroll
  for (int i = 0; i < 4; ++i)
#pragma unroll
    for (int j = 0; j < 4; ++j) acc[i][j] = vz;

  for (int k0 = 0; k0 < K; k0 += 32) {
    ABFrag a[4];
#pragma unroll
    for (int i = 0; i < 4; ++i) {
      const bf16* ap = A + (tileM + 16 * i + l16) * (long)K + k0 + kbA;
      a[i].h[0] = *(const v8bf*)ap;
      a[i].h[1] = *(const v8bf*)(ap + 8);
    }
#pragma unroll
    for (int j = 0; j < 4; ++j) {
      const bf16* bp = W + (tileN + 16 * j + l16) * (long)K + k0 + kbB;
      v16bf bfrag = *(const v16bf*)bp;
      __builtin_prefetch(bp + 32, 0, 1);  // global_prefetch_b8 next K tile
#pragma unroll
      for (int i = 0; i < 4; ++i)
        acc[i][j] = wmma_bf16(a[i].v, bfrag, acc[i][j]);
    }
  }

  // Epilogue: C element (VGPR r, lane) -> row = 16i + r + 8*half, col = 16j + l16
#pragma unroll
  for (int j = 0; j < 4; ++j) {
    const long n = tileN + 16 * j + l16;
    const float bv = bias[n];
#pragma unroll
    for (int i = 0; i < 4; ++i) {
#pragma unroll
      for (int r = 0; r < 8; ++r) {
        const long m = tileM + 16 * i + r + 8 * half;
        const float c = acc[i][j][r] + bv;
        if (mode == 2) {
          ((float*)outp)[m * (long)N + n] = c;
        } else {
          const int b = (int)(m >> 11);   // / 2048
          const int s = (int)(m & 2047);
          const int h = (int)(n >> 7);    // / 128
          const int d = (int)(n & 127);
          long off;
          if (mode == 0)
            off = (((long)(b * 16 + h) * 2048) + s) * 128 + d;   // [b,h,s,d]
          else
            off = (((long)(b * 16 + h) * 128) + d) * 2048 + s;   // [b,h,d,s]
          ((bf16*)outp)[off] = f2bf(c);
        }
      }
    }
  }
}

// ---------------------------------------------------------------------------
// In-place L2-normalize rows of 128 bf16 (one wave per row); optional scale.
// ---------------------------------------------------------------------------
__global__ void __launch_bounds__(256)
norm_rows(bf16* __restrict__ q, const float* __restrict__ scalep,
          int useScale, long nrows) {
  const long row = (long)blockIdx.x * 8 + (threadIdx.x >> 5);
  if (row >= nrows) return;
  const int lane = threadIdx.x & 31;
  bf16* p = q + row * 128 + lane * 4;
  float x[4];
  float ss = 0.f;
#pragma unroll
  for (int t = 0; t < 4; ++t) { x[t] = bf2f(p[t]); ss += x[t] * x[t]; }
  ss += __shfl_xor(ss, 1);
  ss += __shfl_xor(ss, 2);
  ss += __shfl_xor(ss, 4);
  ss += __shfl_xor(ss, 8);
  ss += __shfl_xor(ss, 16);
  const float nrm = sqrtf(ss);
  float inv = 1.0f / fmaxf(nrm, 1e-12f);
  if (useScale) inv *= scalep[0];
#pragma unroll
  for (int t = 0; t < 4; ++t) p[t] = f2bf(x[t] * inv);
}

// ---------------------------------------------------------------------------
// Flash attention: one wave per (b, h, 16-row q chunk).
// Qn/Kn: [b,h,s,128] bf16 (q pre-scaled & normalized). Vt: [b,h,128,s] bf16.
// mask: int64 [b, s]; key masked where 0. ctx out: [b, s, h*128+d] bf16.
// ---------------------------------------------------------------------------
__global__ void __launch_bounds__(256)
flash_attn(const bf16* __restrict__ Qn, const bf16* __restrict__ Kn,
           const bf16* __restrict__ Vt, const long long* __restrict__ mask,
           bf16* __restrict__ ctx) {
  __shared__ __align__(32) bf16 psh[8][16][32];   // per-wave P transpose tile

  const int lane = threadIdx.x & 31;
  const int w    = threadIdx.x >> 5;
  const long gwave = (long)blockIdx.x * 8 + w;    // 0..4095
  const int qc = (int)(gwave & 127);              // 128 q-chunks of 16
  const int bh = (int)(gwave >> 7);               // 0..31
  const int b  = bh >> 4;
  const int h  = bh & 15;
  const int q0 = qc << 4;
  const int half = lane >> 4;
  const int l16  = lane & 15;
  const int kbA  = half << 4;
  const int kbB  = half << 4;

  const bf16* Qb = Qn + (long)bh * 2048 * 128;
  const bf16* Kb = Kn + (long)bh * 2048 * 128;
  const bf16* Vb = Vt + (long)bh * 128 * 2048;
  const long long* mk = mask + (long)b * 2048;

  // Q A-fragments for the 4 K(=d)-groups of 32
  ABFrag aq[4];
#pragma unroll
  for (int g = 0; g < 4; ++g) {
    const bf16* qp = Qb + (long)(q0 + l16) * 128 + g * 32 + kbA;
    aq[g].h[0] = *(const v8bf*)qp;
    aq[g].h[1] = *(const v8bf*)(qp + 8);
  }

  float mi[8], li[8];
  v8f acc[8];
  const v8f vz = {};
#pragma unroll
  for (int r = 0; r < 8; ++r) { mi[r] = NEG_BIG; li[r] = 0.f; }
#pragma unroll
  for (int c = 0; c < 8; ++c) acc[c] = vz;

  for (int k0 = 0; k0 < 2048; k0 += 32) {
    const int col0 = k0 + l16;
    const int col1 = k0 + 16 + l16;

    // S = Q @ K^T : two 16x16 chunks, accumulate over d=128
    v8f s0 = vz, s1 = vz;
#pragma unroll
    for (int g = 0; g < 4; ++g) {
      v16bf bk0 = *(const v16bf*)(Kb + (long)col0 * 128 + g * 32 + kbB);
      v16bf bk1 = *(const v16bf*)(Kb + (long)col1 * 128 + g * 32 + kbB);
      s0 = wmma_bf16(aq[g].v, bk0, s0);
      s1 = wmma_bf16(aq[g].v, bk1, s1);
    }

    const bool live0 = (mk[col0] != 0);
    const bool live1 = (mk[col1] != 0);

    // Online softmax (per-row stats; row = r + 8*half, cols across 16-lane half)
    float p0[8], p1[8];
#pragma unroll
    for (int r = 0; r < 8; ++r) {
      const float a0 = live0 ? s0[r] : NEG_BIG;
      const float a1 = live1 ? s1[r] : NEG_BIG;
      float mx = fmaxf(a0, a1);
      mx = fmaxf(mx, __shfl_xor(mx, 1));
      mx = fmaxf(mx, __shfl_xor(mx, 2));
      mx = fmaxf(mx, __shfl_xor(mx, 4));
      mx = fmaxf(mx, __shfl_xor(mx, 8));
      const float mn = fmaxf(mi[r], mx);
      const float alpha = __expf(mi[r] - mn);
      const float e0 = __expf(a0 - mn);
      const float e1 = __expf(a1 - mn);
      float rs = e0 + e1;
      rs += __shfl_xor(rs, 1);
      rs += __shfl_xor(rs, 2);
      rs += __shfl_xor(rs, 4);
      rs += __shfl_xor(rs, 8);
      li[r] = li[r] * alpha + rs;
      mi[r] = mn;
      p0[r] = e0;
      p1[r] = e1;
#pragma unroll
      for (int c = 0; c < 8; ++c) acc[c][r] *= alpha;
    }

    // P (C-layout) -> LDS -> A-layout fragment
#pragma unroll
    for (int r = 0; r < 8; ++r) {
      psh[w][r + 8 * half][l16]      = f2bf(p0[r]);
      psh[w][r + 8 * half][16 + l16] = f2bf(p1[r]);
    }
    ABFrag pa;
    {
      const bf16* pp = &psh[w][l16][0];
      pa.h[0] = *(const v8bf*)(pp + kbA);
      pa.h[1] = *(const v8bf*)(pp + 8 + kbA);
    }

    // O += P @ V : 8 output chunks of 16 along d, K=32 keys per WMMA
#pragma unroll
    for (int c = 0; c < 8; ++c) {
      v16bf bv = *(const v16bf*)(Vb + (long)(16 * c + l16) * 2048 + k0 + kbB);
      acc[c] = wmma_bf16(pa.v, bv, acc[c]);
    }
  }

  // Epilogue: divide by li, write ctx[b, s, h*128 + d]
#pragma unroll
  for (int r = 0; r < 8; ++r) {
    const float inv = (li[r] > 1e-30f) ? (1.0f / li[r]) : 0.f;
    const long s = q0 + r + 8 * half;
#pragma unroll
    for (int c = 0; c < 8; ++c) {
      ctx[((long)b * 2048 + s) * 2048 + h * 128 + 16 * c + l16] =
          f2bf(acc[c][r] * inv);
    }
  }
}

// ---------------------------------------------------------------------------
extern "C" void kernel_launch(void* const* d_in, const int* in_sizes, int n_in,
                              void* d_out, int out_size, void* d_ws,
                              size_t ws_size, hipStream_t stream) {
  (void)in_sizes; (void)n_in; (void)out_size; (void)ws_size;

  const float* input     = (const float*)d_in[0];
  const long long* mask  = (const long long*)d_in[1];
  const float* Wq = (const float*)d_in[2];
  const float* bq = (const float*)d_in[3];
  const float* Wk = (const float*)d_in[4];
  const float* bk = (const float*)d_in[5];
  const float* Wv = (const float*)d_in[6];
  const float* bv = (const float*)d_in[7];
  const float* Wo = (const float*)d_in[8];
  const float* bo = (const float*)d_in[9];
  const float* att_scale = (const float*)d_in[10];

  const long BS = 2, S = 2048, D = 2048, H = 16;
  const long Mrows = BS * S;  // 4096

  char* p = (char*)d_ws;
  bf16* inB = (bf16*)p; p += Mrows * D * sizeof(bf16);
  bf16* wqB = (bf16*)p; p += D * D * sizeof(bf16);
  bf16* wkB = (bf16*)p; p += D * D * sizeof(bf16);
  bf16* wvB = (bf16*)p; p += D * D * sizeof(bf16);
  bf16* woB = (bf16*)p; p += D * D * sizeof(bf16);
  bf16* Qn  = (bf16*)p; p += Mrows * D * sizeof(bf16);
  bf16* Kn  = (bf16*)p; p += Mrows * D * sizeof(bf16);
  bf16* Vt  = (bf16*)p; p += Mrows * D * sizeof(bf16);
  bf16* ctx = (bf16*)p; p += Mrows * D * sizeof(bf16);

  cvt_f32_to_bf16<<<4096, 256, 0, stream>>>(input, inB, Mrows * D);
  cvt_f32_to_bf16<<<2048, 256, 0, stream>>>(Wq, wqB, D * D);
  cvt_f32_to_bf16<<<2048, 256, 0, stream>>>(Wk, wkB, D * D);
  cvt_f32_to_bf16<<<2048, 256, 0, stream>>>(Wv, wvB, D * D);
  cvt_f32_to_bf16<<<2048, 256, 0, stream>>>(Wo, woB, D * D);

  dim3 ggrid((unsigned)(D / 256), (unsigned)(Mrows / 128));  // (8, 32)
  gemm_wmma_bf16<<<ggrid, 256, 0, stream>>>(inB, wqB, bq, (void*)Qn,
                                            (int)Mrows, (int)D, (int)D, 0);
  gemm_wmma_bf16<<<ggrid, 256, 0, stream>>>(inB, wkB, bk, (void*)Kn,
                                            (int)Mrows, (int)D, (int)D, 0);
  gemm_wmma_bf16<<<ggrid, 256, 0, stream>>>(inB, wvB, bv, (void*)Vt,
                                            (int)Mrows, (int)D, (int)D, 1);

  const long nrows = BS * H * S;  // 65536 head-rows of 128
  norm_rows<<<(unsigned)(nrows / 8), 256, 0, stream>>>(Qn, att_scale, 1, nrows);
  norm_rows<<<(unsigned)(nrows / 8), 256, 0, stream>>>(Kn, att_scale, 0, nrows);

  flash_attn<<<512, 256, 0, stream>>>(Qn, Kn, Vt, mask, ctx);

  gemm_wmma_bf16<<<ggrid, 256, 0, stream>>>(ctx, woB, bo, d_out,
                                            (int)Mrows, (int)D, (int)D, 2);
}
